// Llama2Model_19825569038918
// MI455X (gfx1250) — matrix operand order repaired and checked by reading, compiled, not run on hardware
//
#include <hip/hip_runtime.h>
#include <hip/hip_bf16.h>

#define B_  2
#define S_  1024
#define D_  1024
#define H_  16
#define HD_ 64
#define F_  4096
#define V_  32000
#define L_  4
#define M_  (B_*S_)

typedef __attribute__((ext_vector_type(16))) __bf16 v16bf;
typedef __attribute__((ext_vector_type(8)))  float  v8f;
typedef int v4i_g __attribute__((vector_size(16)));   // matches builtin proto

// ---------------------------------------------------------------------------
// CDNA5 async global->LDS copy (DMA path, tracked by ASYNCcnt)
// ---------------------------------------------------------------------------
#if __has_builtin(__builtin_amdgcn_global_load_async_to_lds_b128)
#define HAVE_ASYNC_LDS 1
#else
#define HAVE_ASYNC_LDS 0
#endif

__device__ __forceinline__ void cp_b128(const __bf16* g, __bf16* l) {
#if HAVE_ASYNC_LDS
    __builtin_amdgcn_global_load_async_to_lds_b128(
        (__attribute__((address_space(1))) v4i_g*)g,
        (__attribute__((address_space(3))) v4i_g*)l,
        0, 0);
#else
    *(uint4*)l = *(const uint4*)g;
#endif
}

__device__ __forceinline__ void wait_async0() {
#if __has_builtin(__builtin_amdgcn_s_wait_asynccnt)
    __builtin_amdgcn_s_wait_asynccnt(0);
#else
    asm volatile("s_wait_asynccnt 0" ::: "memory");
#endif
}

// ---------------------------------------------------------------------------
// Elementwise / reshape kernels
// ---------------------------------------------------------------------------

// Tiled fp32[K,N] -> bf16[N,K] transpose+convert (per-layer via blockIdx.z)
__global__ __launch_bounds__(256) void cvt_t_kernel(const float* __restrict__ src,
                                                    __bf16* __restrict__ dst,
                                                    int Kd, int Nd) {
    __shared__ float tile[32][33];
    size_t zo = (size_t)blockIdx.z * Kd * Nd;
    int kt = blockIdx.x * 32, nt = blockIdx.y * 32;
    int tx = threadIdx.x & 31, ty = threadIdx.x >> 5;   // 32 x 8
    for (int i = ty; i < 32; i += 8)
        tile[i][tx] = src[zo + (size_t)(kt + i) * Nd + nt + tx];
    __syncthreads();
    for (int i = ty; i < 32; i += 8)
        dst[zo + (size_t)(nt + i) * Kd + kt + tx] = (__bf16)tile[tx][i];
}

__global__ __launch_bounds__(256) void embed_kernel(const int* __restrict__ idx,
                                                    const float* __restrict__ emb,
                                                    float* __restrict__ x) {
    int row = blockIdx.x;                      // 0 .. M_-1  (b*S + s)
    int tok = idx[row];
    const float* s = emb + (size_t)tok * D_;
    float*       d = x   + (size_t)row * D_;
    for (int i = threadIdx.x; i < D_; i += 256) d[i] = s[i];
}

__global__ __launch_bounds__(256) void rmsnorm_kernel(const float* __restrict__ x,
                                                      const float* __restrict__ w,
                                                      __bf16* __restrict__ out) {
    __shared__ float red[256];
    int row = blockIdx.x;
    const float* xr = x + (size_t)row * D_;
    float s = 0.f;
    for (int i = threadIdx.x; i < D_; i += 256) { float v = xr[i]; s += v * v; }
    red[threadIdx.x] = s;
    __syncthreads();
    for (int off = 128; off > 0; off >>= 1) {
        if (threadIdx.x < off) red[threadIdx.x] += red[threadIdx.x + off];
        __syncthreads();
    }
    float inv = rsqrtf(red[0] * (1.f / D_) + 1e-5f);
    for (int i = threadIdx.x; i < D_; i += 256)
        out[(size_t)row * D_ + i] = (__bf16)(xr[i] * inv * w[i]);
}

// in: [M_, D_] fp32 -> out: [B_,H_,S_,HD_] bf16 with RoPE applied (for Q,K)
__global__ __launch_bounds__(256) void rope_split_kernel(const float* __restrict__ in,
                                                         __bf16* __restrict__ out) {
    int idx = blockIdx.x * 256 + threadIdx.x;   // ((b*H + h)*S + s)*HD + d
    int d = idx & (HD_ - 1);
    int s = (idx >> 6) & (S_ - 1);
    int h = (idx >> 16) & (H_ - 1);
    int b = idx >> 20;
    size_t base = (size_t)(b * S_ + s) * D_ + h * HD_;
    float v = in[base + d];
    int i = d & 31;
    float freq = powf(10000.f, -(float)(2 * i) / (float)HD_);
    float ang = (float)s * freq;
    float c = cosf(ang), sn = sinf(ang);
    float rot = (d < 32) ? -in[base + d + 32] : in[base + d - 32];
    out[idx] = (__bf16)(v * c + rot * sn);
}

// in: [M_, D_] fp32 -> out: [B_,H_,HD_,S_] bf16 (V stored transposed per head)
__global__ __launch_bounds__(256) void split_vT_kernel(const float* __restrict__ in,
                                                       __bf16* __restrict__ out) {
    int idx = blockIdx.x * 256 + threadIdx.x;   // ((b*H + h)*HD + d)*S + s
    int s = idx & (S_ - 1);
    int d = (idx >> 10) & (HD_ - 1);
    int h = (idx >> 16) & (H_ - 1);
    int b = idx >> 20;
    out[idx] = (__bf16)in[(size_t)(b * S_ + s) * D_ + h * HD_ + d];
}

// ctx: [B_,H_,S_,HD_] fp32 -> out: [M_,D_] bf16 (heads merged)
__global__ __launch_bounds__(256) void merge_ctx_kernel(const float* __restrict__ ctx,
                                                        __bf16* __restrict__ out) {
    int idx = blockIdx.x * 256 + threadIdx.x;   // ((b*S + s)*D) + h*HD + d
    int dd = idx & (HD_ - 1);
    int h  = (idx >> 6) & (H_ - 1);
    int s  = (idx >> 10) & (S_ - 1);
    int b  = idx >> 20;
    size_t src = (((size_t)(b * H_ + h) * S_) + s) * HD_ + dd;
    out[idx] = (__bf16)ctx[src];
}

// Causal masked softmax over one score row; scale 1/sqrt(HD)=0.125 folded in.
// Reads fp32 row, writes bf16 probs IN PLACE at the row base (pitch 2*S_ bf16).
__global__ __launch_bounds__(256) void softmax_kernel(float* __restrict__ scores) {
    __shared__ float red[256];
    int i  = blockIdx.x;            // query position
    int bh = blockIdx.y;
    float* rowp = scores + ((size_t)bh * S_ + i) * S_;
    int tid = threadIdx.x;

    float vals[4];
    float mx = -3.0e38f;
    #pragma unroll
    for (int t = 0; t < 4; ++t) {
        int j = tid + t * 256;
        float v = (j <= i) ? rowp[j] * 0.125f : -3.0e38f;
        vals[t] = v;
        mx = fmaxf(mx, v);
    }
    red[tid] = mx; __syncthreads();
    for (int off = 128; off > 0; off >>= 1) {
        if (tid < off) red[tid] = fmaxf(red[tid], red[tid + off]);
        __syncthreads();
    }
    mx = red[0];
    __syncthreads();

    float sum = 0.f;
    #pragma unroll
    for (int t = 0; t < 4; ++t) {
        int j = tid + t * 256;
        float e = (j <= i) ? __expf(vals[t] - mx) : 0.f;
        vals[t] = e;
        sum += e;
    }
    red[tid] = sum; __syncthreads();
    for (int off = 128; off > 0; off >>= 1) {
        if (tid < off) red[tid] += red[tid + off];
        __syncthreads();
    }
    float inv = 1.f / red[0];

    __bf16* op = (__bf16*)rowp;
    #pragma unroll
    for (int t = 0; t < 4; ++t) {
        int j = tid + t * 256;
        op[j] = (__bf16)(vals[t] * inv);
    }
}

__global__ __launch_bounds__(256) void silu_mul_kernel(const float* __restrict__ a,
                                                       const float* __restrict__ b,
                                                       __bf16* __restrict__ g) {
    long long i = (long long)blockIdx.x * 256 + threadIdx.x;
    float x = a[i];
    float si = x / (1.f + __expf(-x));
    g[i] = (__bf16)(si * b[i]);
}

// ---------------------------------------------------------------------------
// Batched bf16 GEMM, f32 accumulation via v_wmma_f32_16x16x32_bf16.
//   C[z] = A[z] (MxK, row-major) * B[z]^T  (B stored [N,K] row-major)
//   (+ bias[N]) (+ res, same layout as C)
// Block: 128 threads = 4 waves in 2x2; each wave owns a 32x32 output
// (2x2 WMMA tiles) -> block tile 64x64, BK=64.
// Double-buffered LDS filled by GLOBAL_LOAD_ASYNC_TO_LDS_B128 (no staging
// VGPRs, DMA overlapped with the WMMA stream), 1 barrier per k-iteration.
// ---------------------------------------------------------------------------
#define BM  64
#define BN  64
#define BK  64
#define LP  72    // LDS row pitch in bf16 elems: 144 B (16B multiple, bank-skewed)
#define NT_ 128

__global__ __launch_bounds__(NT_, 1) void gemm_bf16_kernel(
    const __bf16* __restrict__ A, int lda, long long sA,
    const __bf16* __restrict__ B, int ldb, long long sB,
    float* __restrict__ C, int ldc, long long sC,
    const float* __restrict__ bias, const float* __restrict__ res,
    int K)
{
    __shared__ __attribute__((aligned(16))) __bf16 As[2][BM * LP];
    __shared__ __attribute__((aligned(16))) __bf16 Bs[2][BN * LP];

    const __bf16* Ab = A + (size_t)blockIdx.z * sA;
    const __bf16* Bb = B + (size_t)blockIdx.z * sB;
    float* Cb = C + (size_t)blockIdx.z * sC;
    const float* Rb = res ? res + (size_t)blockIdx.z * sC : (const float*)0;

    int tid  = threadIdx.x;
    int lane = tid & 31;
    int wave = tid >> 5;        // 0..3
    int wm   = wave >> 1;       // 0..1
    int wn   = wave & 1;        // 0..1
    int m0   = blockIdx.x * BM;
    int n0   = blockIdx.y * BN;

    int lrow = lane & 15;
    int hh   = lane >> 4;       // lane half
    int akb  = hh * 8;          // A frag: K = akb..+7 and 16+akb..+7 (interleaved)
    int bkb  = hh * 8;          // B frag: K = bkb..+7 and 16+bkb..+7 (interleaved)

    v8f zero = {0.f, 0.f, 0.f, 0.f, 0.f, 0.f, 0.f, 0.f};
    v8f acc[2][2] = {{zero, zero}, {zero, zero}};

    // Per-thread staging slices: 4 x 16B for A and for B per tile.
    auto stage = [&](int k0, int buf) {
        #pragma unroll
        for (int it = 0; it < 4; ++it) {
            int e = (tid + it * NT_) * 8;
            int r = e >> 6, c = e & 63;
            cp_b128(Ab + (size_t)(m0 + r) * lda + (k0 + c), &As[buf][r * LP + c]);
            cp_b128(Bb + (size_t)(n0 + r) * ldb + (k0 + c), &Bs[buf][r * LP + c]);
        }
    };

    stage(0, 0);
    wait_async0();
    __syncthreads();

    int nIter = K / BK;
    for (int i = 0; i < nIter; ++i) {
        int buf = i & 1;
        if (i + 1 < nIter) stage((i + 1) * BK, buf ^ 1);   // DMA overlapped w/ WMMA

        #pragma unroll
        for (int kc = 0; kc < BK; kc += 32) {
            union { v16bf v; uint4 q[2]; } af[2], bfq[2];
            #pragma unroll
            for (int mt = 0; mt < 2; ++mt) {
                const __bf16* s = &As[buf][(wm * 32 + mt * 16 + lrow) * LP + kc + akb];
                af[mt].q[0] = *(const uint4*)s;
                af[mt].q[1] = *(const uint4*)(s + 16);
            }
            #pragma unroll
            for (int nt = 0; nt < 2; ++nt) {
                const __bf16* s = &Bs[buf][(wn * 32 + nt * 16 + lrow) * LP + kc + bkb];
                bfq[nt].q[0] = *(const uint4*)s;
                bfq[nt].q[1] = *(const uint4*)(s + 16);
            }
            #pragma unroll
            for (int mt = 0; mt < 2; ++mt)
                #pragma unroll
                for (int nt = 0; nt < 2; ++nt)
                    acc[mt][nt] = __builtin_amdgcn_wmma_f32_16x16x32_bf16(
                        false, af[mt].v, false, bfq[nt].v, (short)0,
                        acc[mt][nt], false, false);
        }

        wait_async0();
        __syncthreads();
    }

    // C/D layout: lane half selects M base (+8), VGPR r -> row r, col = lane&15
    int col = lane & 15;
    #pragma unroll
    for (int mt = 0; mt < 2; ++mt) {
        #pragma unroll
        for (int nt = 0; nt < 2; ++nt) {
            int gm0 = m0 + wm * 32 + mt * 16 + hh * 8;
            int gn  = n0 + wn * 32 + nt * 16 + col;
            float bv = bias ? bias[gn] : 0.f;
            #pragma unroll
            for (int r = 0; r < 8; ++r) {
                size_t o = (size_t)(gm0 + r) * ldc + gn;
                float v = acc[mt][nt][r] + bv;
                if (Rb) v += Rb[o];
                Cb[o] = v;
            }
        }
    }
}

// ---------------------------------------------------------------------------
// Host orchestration
// ---------------------------------------------------------------------------
extern "C" void kernel_launch(void* const* d_in, const int* in_sizes, int n_in,
                              void* d_out, int out_size, void* d_ws, size_t ws_size,
                              hipStream_t stream) {
    (void)in_sizes; (void)n_in; (void)out_size; (void)ws_size;

    const int*   in_idx  = (const int*)  d_in[0];
    const float* tok_emb = (const float*)d_in[1];
    const float* Wq      = (const float*)d_in[2];
    const float* Wk      = (const float*)d_in[3];
    const float* Wv      = (const float*)d_in[4];
    const float* Wo      = (const float*)d_in[5];
    const float* bo      = (const float*)d_in[6];
    const float* fc1     = (const float*)d_in[7];
    const float* fc2     = (const float*)d_in[8];
    const float* fc3     = (const float*)d_in[9];
    const float* n1      = (const float*)d_in[10];
    const float* n2      = (const float*)d_in[11];
    const float* fw      = (const float*)d_in[12];
    const float* outW    = (const float*)d_in[13];
    float* logits = (float*)d_out;

    char* p = (char*)d_ws;
    size_t off = 0;
    auto alloc = [&](size_t bytes) -> void* {
        off = (off + 255) & ~(size_t)255;
        void* r = p + off;
        off += bytes;
        return r;
    };

    // All weights stored bf16 TRANSPOSED: [N,K] row-major
    __bf16* wq_t = (__bf16*)alloc((size_t)L_ * D_ * D_ * 2);
    __bf16* wk_t = (__bf16*)alloc((size_t)L_ * D_ * D_ * 2);
    __bf16* wv_t = (__bf16*)alloc((size_t)L_ * D_ * D_ * 2);
    __bf16* wo_t = (__bf16*)alloc((size_t)L_ * D_ * D_ * 2);
    __bf16* f1_t = (__bf16*)alloc((size_t)L_ * D_ * F_ * 2);   // [F,D] per layer
    __bf16* f2_t = (__bf16*)alloc((size_t)L_ * D_ * F_ * 2);   // [F,D] per layer
    __bf16* f3_t = (__bf16*)alloc((size_t)L_ * F_ * D_ * 2);   // [D,F] per layer
    __bf16* ow_t = (__bf16*)alloc((size_t)D_ * V_ * 2);        // [V,D]
    float*  x    = (float*) alloc((size_t)M_ * D_ * 4);
    __bf16* hbuf = (__bf16*)alloc((size_t)M_ * D_ * 2);
    __bf16* qbf  = (__bf16*)alloc((size_t)M_ * D_ * 2);        // [B,H,S,HD]
    __bf16* kbf  = (__bf16*)alloc((size_t)M_ * D_ * 2);        // [B,H,S,HD]
    __bf16* vtb  = (__bf16*)alloc((size_t)M_ * D_ * 2);        // [B,H,HD,S]
    __bf16* ctxm = (__bf16*)alloc((size_t)M_ * D_ * 2);
    float*  sfa  = (float*) alloc((size_t)M_ * F_ * 4);
    float*  scores = (float*)alloc((size_t)B_ * H_ * S_ * S_ * 4);
    // FFN-phase scratch aliased into the (attention-phase-only) scores buffer:
    float*  sfb  = scores;                                          // 32 MB
    __bf16* gbuf = (__bf16*)((char*)scores + (size_t)M_ * F_ * 4);  // +16 MB

    auto cvtT = [&](const float* s, __bf16* d, int Kd, int Nd, int layers) {
        cvt_t_kernel<<<dim3(Kd / 32, Nd / 32, layers), 256, 0, stream>>>(s, d, Kd, Nd);
    };
    cvtT(Wq,  wq_t, D_, D_, L_);
    cvtT(Wk,  wk_t, D_, D_, L_);
    cvtT(Wv,  wv_t, D_, D_, L_);
    cvtT(Wo,  wo_t, D_, D_, L_);
    cvtT(fc1, f1_t, D_, F_, L_);
    cvtT(fc2, f2_t, D_, F_, L_);
    cvtT(fc3, f3_t, F_, D_, L_);
    cvtT(outW, ow_t, D_, V_, 1);

    auto gemm = [&](const __bf16* A, int lda, long long sA,
                    const __bf16* Bp, int ldb, long long sB,
                    float* C, int ldc, long long sC,
                    const float* bias, const float* res,
                    int M, int N, int K, int batch) {
        gemm_bf16_kernel<<<dim3(M / BM, N / BN, batch), NT_, 0, stream>>>(
            A, lda, sA, Bp, ldb, sB, C, ldc, sC, bias, res, K);
    };

    embed_kernel<<<M_, 256, 0, stream>>>(in_idx, tok_emb, x);

    const int eltBlocks = (M_ * D_) / 256;   // 8192

    for (int l = 0; l < L_; ++l) {
        rmsnorm_kernel<<<M_, 256, 0, stream>>>(x, n1 + (size_t)l * D_, hbuf);

        // Q, K, V projections (+RoPE for Q,K); V written transposed per head
        gemm(hbuf, D_, 0, wq_t + (size_t)l * D_ * D_, D_, 0,
             sfa, D_, 0, nullptr, nullptr, M_, D_, D_, 1);
        rope_split_kernel<<<eltBlocks, 256, 0, stream>>>(sfa, qbf);

        gemm(hbuf, D_, 0, wk_t + (size_t)l * D_ * D_, D_, 0,
             sfa, D_, 0, nullptr, nullptr, M_, D_, D_, 1);
        rope_split_kernel<<<eltBlocks, 256, 0, stream>>>(sfa, kbf);

        gemm(hbuf, D_, 0, wv_t + (size_t)l * D_ * D_, D_, 0,
             sfa, D_, 0, nullptr, nullptr, M_, D_, D_, 1);
        split_vT_kernel<<<eltBlocks, 256, 0, stream>>>(sfa, vtb);

        // scores[bh] = q[bh] @ k[bh]^T   (k stored [S,HD] = [N,K])
        gemm(qbf, HD_, (long long)S_ * HD_,
             kbf, HD_, (long long)S_ * HD_,
             scores, S_, (long long)S_ * S_,
             nullptr, nullptr, S_, S_, HD_, B_ * H_);

        softmax_kernel<<<dim3(S_, B_ * H_), 256, 0, stream>>>(scores);

        // ctx[bh] = probs[bh] @ v[bh]   (v stored [HD,S] = [N,K]; probs bf16
        // in-place over the f32 scores rows -> pitch 2*S_)
        gemm((const __bf16*)scores, 2 * S_, (long long)2 * S_ * S_,
             vtb, S_, (long long)HD_ * S_,
             sfa, HD_, (long long)S_ * HD_,
             nullptr, nullptr, S_, HD_, S_, B_ * H_);

        merge_ctx_kernel<<<eltBlocks, 256, 0, stream>>>(sfa, ctxm);

        // x = x + ctx @ Wo + bo
        gemm(ctxm, D_, 0, wo_t + (size_t)l * D_ * D_, D_, 0,
             x, D_, 0, bo + (size_t)l * D_, x, M_, D_, D_, 1);

        // FFN: x = x + (silu(h@fc1) * (h@fc2)) @ fc3
        rmsnorm_kernel<<<M_, 256, 0, stream>>>(x, n2 + (size_t)l * D_, hbuf);

        gemm(hbuf, D_, 0, f1_t + (size_t)l * D_ * F_, D_, 0,
             sfa, F_, 0, nullptr, nullptr, M_, F_, D_, 1);
        gemm(hbuf, D_, 0, f2_t + (size_t)l * D_ * F_, D_, 0,
             sfb, F_, 0, nullptr, nullptr, M_, F_, D_, 1);
        silu_mul_kernel<<<(M_ * F_) / 256, 256, 0, stream>>>(sfa, sfb, gbuf);
        gemm(gbuf, F_, 0, f3_t + (size_t)l * F_ * D_, F_, 0,
             x, D_, 0, nullptr, x, M_, D_, F_, 1);
    }

    rmsnorm_kernel<<<M_, 256, 0, stream>>>(x, fw, hbuf);
    gemm(hbuf, D_, 0, ow_t, D_, 0,
         logits, V_, 0, nullptr, nullptr, M_, V_, D_, 1);
}
